// ScaledDotProductAttention_6992206758194
// MI455X (gfx1250) — matrix operand order repaired and checked by reading, not compiled, not run on hardware
//
#include <hip/hip_runtime.h>
#include <hip/hip_fp16.h>

typedef __attribute__((ext_vector_type(16))) _Float16 v16h;
typedef __attribute__((ext_vector_type(8)))  float    v8f;
typedef __attribute__((ext_vector_type(4)))  unsigned int u32x4;
typedef __attribute__((ext_vector_type(8)))  int      i32x8;
typedef __attribute__((ext_vector_type(4)))  int      i32x4;

struct __align__(16) U4 { unsigned int x, y, z, w; };

union Frag16 {
    v16h     v;
    _Float16 h[16];
    U4       q[2];
};

static __device__ __forceinline__ v8f wmma_f16(v16h a, v16h b, v8f c) {
    // D = A(16x32 f16) * B(32x16 f16) + C(16x16 f32)
    return __builtin_amdgcn_wmma_f32_16x16x32_f16(
        /*neg_a=*/false, a, /*neg_b=*/false, b,
        /*c_mod=*/(short)0, c, /*reuse_a=*/false, /*reuse_b=*/false);
}

// Problem constants (fixed by the reference).
constexpr int kB  = 4;
constexpr int kT  = 4096;   // power of two
constexpr int kC  = 1024;
constexpr int kHD = 64;
constexpr int kTShift = 12;        // log2(kT)
constexpr int kTMask  = kT - 1;

// ---------------------------------------------------------------------------
// TDM: issue a 2-D f16 tile load Global -> LDS (D# per CDNA5 ISA ch.8).
//   tile_d0 x tile_d1 elements, source row stride = stride_d0 (elements),
//   destination LDS tile is packed (tile_d0 elements per row).
// ---------------------------------------------------------------------------
static __device__ __forceinline__ void tdm_load_2d_f16(
    unsigned lds_off, const void* gptr,
    unsigned tensor_d0, unsigned tensor_d1,
    unsigned tile_d0, unsigned tile_d1, unsigned stride_d0) {
    const unsigned long long ga = (unsigned long long)(__SIZE_TYPE__)gptr;
    u32x4 g0;
    g0[0] = 1u;                                     // count=1, user descriptor
    g0[1] = lds_off;                                // lds_addr (bytes)
    g0[2] = (unsigned)ga;                           // global_addr[31:0]
    g0[3] = (unsigned)(ga >> 32) | (2u << 30);      // global_addr[56:32] | type=2
    i32x8 g1;
    g1[0] = (int)(1u << 16);                        // data_size=1 (2B); wg_mask=0
    g1[1] = (int)((tensor_d0 & 0xFFFFu) << 16);     // tensor_dim0[15:0]
    g1[2] = (int)(((tensor_d0 >> 16) & 0xFFFFu) | ((tensor_d1 & 0xFFFFu) << 16));
    g1[3] = (int)(((tensor_d1 >> 16) & 0xFFFFu) | ((tile_d0 & 0xFFFFu) << 16));
    g1[4] = (int)(tile_d1 & 0xFFFFu);               // tile_dim1; tile_dim2=0
    g1[5] = (int)stride_d0;                         // tensor_dim0_stride[31:0]
    g1[6] = 0;                                      // stride[47:32], dim1_stride lo
    g1[7] = 0;
    const i32x4 z4 = {0, 0, 0, 0};                  // groups 2/3 unused (2-D)
    const i32x8 z8 = {0, 0, 0, 0, 0, 0, 0, 0};
    __builtin_amdgcn_tensor_load_to_lds(g0, g1, z4, z4, z8, 0);
}

// ---------------------------------------------------------------------------
// Projection: OUT[row, col] = sum_c X[row, c] * W[col, c]   (f32 in, f16 out)
// Block = 128 threads (4 waves); 16 rows x 64 cols per block, 16x16 per wave.
// TRANSPOSED==0: OUT is [NROWS, 64] f16. TRANSPOSED==1: OUT is [B, 64, T] f16.
// ---------------------------------------------------------------------------
template <int TRANSPOSED>
__global__ __launch_bounds__(128) void proj_kernel(
    const float* __restrict__ X, const float* __restrict__ W,
    _Float16* __restrict__ out, float scale) {
    const int lane  = threadIdx.x & 31;
    const int wave  = threadIdx.x >> 5;
    const int row0  = blockIdx.x * 16;
    const int col0  = wave * 16;
    const int mrow  = lane & 15;
    const int koffA = (lane < 16) ? 0 : 8;   // A-fragment K interleave
    const int koffB = (lane < 16) ? 0 : 8;   // B-fragment K interleave

    const float* xrow = X + (size_t)(row0 + mrow) * kC;
    const float* wrow = W + (size_t)(col0 + mrow) * kC;

    v8f acc = {};
    for (int k0 = 0; k0 < kC; k0 += 32) {
        Frag16 a, b;
        const float* xa0 = xrow + k0 + koffA;
        const float* xa1 = xrow + k0 + 16 + koffA;
        const float* wb  = wrow + k0 + koffB;
#pragma unroll
        for (int e = 0; e < 8; ++e) {
            a.h[e]     = (_Float16)xa0[e];
            a.h[8 + e] = (_Float16)xa1[e];
            b.h[e]     = (_Float16)wb[e];
            b.h[8 + e] = (_Float16)wb[16 + e];
        }
        acc = wmma_f16(a.v, b.v, acc);
    }

    const int rbase = (lane < 16) ? 0 : 8;
#pragma unroll
    for (int r = 0; r < 8; ++r) {
        const int grow = row0 + rbase + r;
        const int gcol = col0 + mrow;
        const _Float16 val = (_Float16)(acc[r] * scale);
        if (TRANSPOSED == 0) {
            out[(size_t)grow * kHD + gcol] = val;
        } else {
            const int bb = grow >> kTShift, tt = grow & kTMask;
            out[((size_t)bb * kHD + gcol) * kT + tt] = val;
        }
    }
}

// ---------------------------------------------------------------------------
// Flash attention, block-cooperative K/V staging via TDM.
// Block = 8 waves, 128 q-rows; wave w owns the 16-row tile at qB + 16w.
// Wave 0 DMAs each 32-s chunk of K (32x64) and V^T (64x32) into
// double-buffered LDS; all waves consume B-fragments from LDS.
// qf/kf: [B,T,64] f16, vt: [B,64,T] f16, out: [B,T,64] fp32.
// ---------------------------------------------------------------------------
__global__ __launch_bounds__(256) void attn_kernel(
    const _Float16* __restrict__ qf, const _Float16* __restrict__ kf,
    const _Float16* __restrict__ vt, float* __restrict__ out) {
    __shared__ __align__(16) _Float16 kbuf[2][32][64];  // K rows    (4KB x2)
    __shared__ __align__(16) _Float16 vbuf[2][64][32];  // V^T rows  (4KB x2)
    __shared__ __align__(16) _Float16 pbuf[8][16][32];  // per-wave P tile

    const int lane  = threadIdx.x & 31;
    const int wave  = threadIdx.x >> 5;
    const int growB = blockIdx.x * 128;          // block's first flat row
    const int bidx  = growB >> kTShift;
    const int qB    = growB & kTMask;            // block's first q within T
    const int q0    = qB + wave * 16;            // this wave's q tile
    const int grow0 = growB + wave * 16;
    const int mrow  = lane & 15;
    const int koffA = (lane < 16) ? 0 : 8;
    const int koffB = (lane < 16) ? 0 : 8;
    const int rbase = (lane < 16) ? 0 : 8;

    const _Float16* kbase = kf + ((size_t)(bidx << kTShift)) * kHD;
    const _Float16* vbat  = vt + (size_t)bidx * kHD * kT;

    // Q A-fragments (16x64 = two 16x32 frags); C^-0.5 folded into qf.
    Frag16 aq0, aq1;
    const _Float16* qrow = qf + ((size_t)(bidx << kTShift) + q0 + mrow) * kHD;
    aq0.q[0] = *reinterpret_cast<const U4*>(qrow + koffA);
    aq0.q[1] = *reinterpret_cast<const U4*>(qrow + 16 + koffA);
    aq1.q[0] = *reinterpret_cast<const U4*>(qrow + 32 + koffA);
    aq1.q[1] = *reinterpret_cast<const U4*>(qrow + 48 + koffA);

    v8f o0 = {}, o1 = {}, o2 = {}, o3 = {};
    float M[8], L[8];
#pragma unroll
    for (int r = 0; r < 8; ++r) { M[r] = -__builtin_inff(); L[r] = 0.0f; }

    const int nc_blk = qB / 32 + 4;              // chunks the block must stage
    const int nc_me  = (q0 + 16 + 31) / 32;      // chunks this wave computes

    // Prologue: stage chunk 0 into buffer 0.
    if (wave == 0) {
        tdm_load_2d_f16((unsigned)(__SIZE_TYPE__)&kbuf[0][0][0], kbase,
                        kHD, kT, kHD, 32, kHD);
        tdm_load_2d_f16((unsigned)(__SIZE_TYPE__)&vbuf[0][0][0], vbat,
                        kT, kHD, 32, kHD, kT);
    }

    for (int c = 0; c < nc_blk; ++c) {
        const int s0  = c * 32;
        const int cur = c & 1;

        if (wave == 0) {
            __builtin_amdgcn_s_wait_tensorcnt(0);  // chunk c resident in LDS
        }
        __syncthreads();                           // publish to all waves

        // Overlap: stage chunk c+1 into the other buffer while computing c.
        if (wave == 0 && c + 1 < nc_blk) {
            const int sn = s0 + 32;
            tdm_load_2d_f16((unsigned)(__SIZE_TYPE__)&kbuf[cur ^ 1][0][0],
                            kbase + (size_t)sn * kHD, kHD, kT, kHD, 32, kHD);
            tdm_load_2d_f16((unsigned)(__SIZE_TYPE__)&vbuf[cur ^ 1][0][0],
                            vbat + sn, kT, kHD, 32, kHD, kT);
        }

        if (c < nc_me) {
            // ---- scores: two 16x16 tiles, K = 64, B-frags from LDS ----
            v8f c0 = {}, c1 = {};
            {
                Frag16 b0, b1, b2, b3;
                const _Float16* kr0 = &kbuf[cur][mrow][0];
                const _Float16* kr1 = &kbuf[cur][16 + mrow][0];
                b0.q[0] = *reinterpret_cast<const U4*>(kr0 + koffB);
                b0.q[1] = *reinterpret_cast<const U4*>(kr0 + koffB + 16);
                b1.q[0] = *reinterpret_cast<const U4*>(kr0 + 32 + koffB);
                b1.q[1] = *reinterpret_cast<const U4*>(kr0 + 32 + koffB + 16);
                b2.q[0] = *reinterpret_cast<const U4*>(kr1 + koffB);
                b2.q[1] = *reinterpret_cast<const U4*>(kr1 + koffB + 16);
                b3.q[0] = *reinterpret_cast<const U4*>(kr1 + 32 + koffB);
                b3.q[1] = *reinterpret_cast<const U4*>(kr1 + 32 + koffB + 16);
                c0 = wmma_f16(aq0.v, b0.v, c0);
                c0 = wmma_f16(aq1.v, b1.v, c0);
                c1 = wmma_f16(aq0.v, b2.v, c1);
                c1 = wmma_f16(aq1.v, b3.v, c1);
            }

            // ---- causal mask (C layout: row = rbase+r, col = mrow) ----
            const int gs0 = s0 + mrow;
#pragma unroll
            for (int r = 0; r < 8; ++r) {
                const int q_abs = q0 + rbase + r;
                if (gs0 > q_abs)      c0[r] = -__builtin_inff();
                if (gs0 + 16 > q_abs) c1[r] = -__builtin_inff();
            }

            // ---- online softmax (reductions within each 16-lane half) ----
            float alpha[8];
#pragma unroll
            for (int r = 0; r < 8; ++r) {
                float tm = fmaxf(c0[r], c1[r]);
#pragma unroll
                for (int off = 1; off < 16; off <<= 1)
                    tm = fmaxf(tm, __shfl_xor(tm, off, 32));
                const float mnew = fmaxf(M[r], tm);
                alpha[r] = __expf(M[r] - mnew);
                M[r] = mnew;
            }
#pragma unroll
            for (int r = 0; r < 8; ++r) {
                const float p0 = __expf(c0[r] - M[r]);
                const float p1 = __expf(c1[r] - M[r]);
                float rs = p0 + p1;
#pragma unroll
                for (int off = 1; off < 16; off <<= 1)
                    rs += __shfl_xor(rs, off, 32);
                L[r] = L[r] * alpha[r] + rs;
                pbuf[wave][rbase + r][mrow]      = (_Float16)p0;
                pbuf[wave][rbase + r][16 + mrow] = (_Float16)p1;
            }
#pragma unroll
            for (int r = 0; r < 8; ++r) {
                o0[r] *= alpha[r]; o1[r] *= alpha[r];
                o2[r] *= alpha[r]; o3[r] *= alpha[r];
            }

            // Wave-private LDS P tile: order stores before A-frag reload.
            asm volatile("s_wait_dscnt 0" ::: "memory");

            // ---- P (A-frag from LDS) x V^T (B-frags from LDS), K = 32 ----
            Frag16 ap;
            ap.q[0] = *reinterpret_cast<const U4*>(&pbuf[wave][mrow][koffA]);
            ap.q[1] = *reinterpret_cast<const U4*>(&pbuf[wave][mrow][16 + koffA]);
#pragma unroll
            for (int f = 0; f < 4; ++f) {
                Frag16 bv;
                const _Float16* vr = &vbuf[cur][f * 16 + mrow][0];
                bv.q[0] = *reinterpret_cast<const U4*>(vr + koffB);
                bv.q[1] = *reinterpret_cast<const U4*>(vr + koffB + 16);
                switch (f) {
                    case 0: o0 = wmma_f16(ap.v, bv.v, o0); break;
                    case 1: o1 = wmma_f16(ap.v, bv.v, o1); break;
                    case 2: o2 = wmma_f16(ap.v, bv.v, o2); break;
                    case 3: o3 = wmma_f16(ap.v, bv.v, o3); break;
                }
            }
        }
        __syncthreads();   // all reads of buffer `cur` done before next overwrite
    }

    // ---- epilogue: divide by row sum, store fp32 ----
#pragma unroll
    for (int r = 0; r < 8; ++r) {
        const float inv = 1.0f / L[r];
        const size_t ro = (size_t)(grow0 + rbase + r) * kHD;
        out[ro + mrow]      = o0[r] * inv;
        out[ro + 16 + mrow] = o1[r] * inv;
        out[ro + 32 + mrow] = o2[r] * inv;
        out[ro + 48 + mrow] = o3[r] * inv;
    }
}

extern "C" void kernel_launch(void* const* d_in, const int* in_sizes, int n_in,
                              void* d_out, int out_size, void* d_ws, size_t ws_size,
                              hipStream_t stream) {
    (void)in_sizes; (void)n_in; (void)out_size; (void)ws_size;
    const float* q_embs = (const float*)d_in[0];
    const float* k_embs = (const float*)d_in[1];
    const float* v_embs = (const float*)d_in[2];
    const float* w_q    = (const float*)d_in[3];
    const float* w_k    = (const float*)d_in[4];
    const float* w_v    = (const float*)d_in[5];
    float* out = (float*)d_out;

    const int N = kB * kT;  // 16384 rows

    _Float16* qf = (_Float16*)d_ws;                  // [B,T,64]
    _Float16* kf = qf + (size_t)N * kHD;             // [B,T,64]
    _Float16* vt = kf + (size_t)N * kHD;             // [B,64,T]

    const float scale = 0.03125f;  // C^-0.5 = 1024^-0.5 = 1/32, folded into Q

    proj_kernel<0><<<N / 16, 128, 0, stream>>>(q_embs, w_q, qf, scale);
    proj_kernel<0><<<N / 16, 128, 0, stream>>>(k_embs, w_k, kf, 1.0f);
    proj_kernel<1><<<N / 16, 128, 0, stream>>>(v_embs, w_v, vt, 1.0f);

    attn_kernel<<<N / 128, 256, 0, stream>>>(qf, kf, vt, out);
}
